// LiquidMemoryCell_49984829390881
// MI455X (gfx1250) — matrix operand order repaired and checked by reading, not compiled, not run on hardware
//
#include <hip/hip_runtime.h>
#include <hip/hip_bf16.h>
#include <math.h>

#define B_ROWS  8192
#define IN_DIM  4096
#define MEM_    256
#define D_IN    4352    // MEM + IN_DIM
#define N_TOTAL 1024    // 512 (W1) + 256 (Wd) + 256 (Wu)

typedef __bf16 bf16_t;
typedef __attribute__((ext_vector_type(16))) __bf16 v16bf;
typedef __attribute__((ext_vector_type(8)))  __bf16 bf16x8;
typedef __attribute__((ext_vector_type(8)))  float  v8f;
typedef unsigned int __attribute__((ext_vector_type(4))) v4u;
typedef int          __attribute__((ext_vector_type(8))) v8i;
typedef int          __attribute__((ext_vector_type(4))) v4i;

#if defined(__has_builtin)
#if __has_builtin(__builtin_amdgcn_tensor_load_to_lds)
#define USE_TDM 1
#endif
#endif

__device__ __forceinline__ float sigm_(float v) { return 1.0f / (1.0f + __expf(-v)); }

// ---- CDNA5 async global->LDS copy (ASYNCcnt path), 16B per lane ----
__device__ __forceinline__ void async_copy16(void* lds_ptr, const void* gptr) {
    unsigned           lo = (unsigned)(uintptr_t)lds_ptr;            // LDS byte offset
    unsigned long long ga = (unsigned long long)(uintptr_t)gptr;
    asm volatile("global_load_async_to_lds_b128 %0, %1, off"
                 :: "v"(lo), "v"(ga) : "memory");
}
__device__ __forceinline__ void wait_async_le4() { asm volatile("s_wait_asynccnt 0x4" ::: "memory"); }
__device__ __forceinline__ void wait_async_le5() { asm volatile("s_wait_asynccnt 0x5" ::: "memory"); }
__device__ __forceinline__ void wait_async_0()   { asm volatile("s_wait_asynccnt 0x0" ::: "memory"); }

#ifdef USE_TDM
// ---- CDNA5 Tensor Data Mover: 2-D tile (tile_d0 x tile_d1, 2B elems) -> LDS ----
// LDS rows padded: 64B data + 16B pad (pad_interval=3 -> 16 DWORDs, pad_amount=3 -> 4 DWORDs)
__device__ __forceinline__ void tdm_load_2d(unsigned lds_off, const void* gaddr,
                                            unsigned tile_d0, unsigned tile_d1,
                                            unsigned tensor_d0, unsigned tensor_d1,
                                            unsigned stride0_elems)
{
    unsigned long long ga = (unsigned long long)(uintptr_t)gaddr;
    v4u g0;
    g0[0] = 1u;                                                  // count=1, user mode
    g0[1] = lds_off;                                             // lds_addr (bytes)
    g0[2] = (unsigned)(ga & 0xFFFFFFFFu);                        // global_addr[31:0]
    g0[3] = (unsigned)((ga >> 32) & 0x01FFFFFFu) | (2u << 30);   // global_addr[56:32], type=2
    v8i g1;
    g1[0] = (int)((1u << 16) | (1u << 20) | (3u << 22) | (3u << 25)); // data_size=2B, pad_en, 64B+16B pad
    g1[1] = (int)((tensor_d0 & 0xFFFFu) << 16);                       // atomic_addr=0 | tensor_dim0[15:0]
    g1[2] = (int)(((tensor_d0 >> 16) & 0xFFFFu) | ((tensor_d1 & 0xFFFFu) << 16));
    g1[3] = (int)(((tensor_d1 >> 16) & 0xFFFFu) | ((tile_d0 & 0xFFFFu) << 16));
    g1[4] = (int)(tile_d1 & 0xFFFFu);                                 // tile_dim1 | tile_dim2=0
    g1[5] = (int)stride0_elems;                                       // tensor_dim0_stride[31:0]
    g1[6] = 0;                                                        // stride0 hi | dim1_stride lo (2-D)
    g1[7] = 0;
    v4i g2 = {0, 0, 0, 0};
    v4i g3 = {0, 0, 0, 0};
    v8i g4 = {0, 0, 0, 0, 0, 0, 0, 0};                                // extra group (6-arg form)
    __builtin_amdgcn_tensor_load_to_lds(g0, g1, g2, g3, g4, 0);
}
__device__ __forceinline__ void wait_tensor_le2() { __builtin_amdgcn_s_wait_tensorcnt((short)2); }
__device__ __forceinline__ void wait_tensor_0()  { __builtin_amdgcn_s_wait_tensorcnt((short)0); }
#endif

// ---------------------------------------------------------------------------
// Pack kernels: fp32 -> bf16, weights pre-transposed so GEMM tiles are
// contiguous 16B chunks (TDM / async-copyable straight into LDS).
// ---------------------------------------------------------------------------
__global__ __launch_bounds__(256)
void lmc_pack_combined(const float* __restrict__ x, const float* __restrict__ h,
                       bf16_t* __restrict__ Cb)
{
    const size_t gid  = (size_t)blockIdx.x * 256 + threadIdx.x;  // 8-elem chunks
    const size_t base = gid * 8;
    const int row = (int)(base / D_IN);
    const int col = (int)(base % D_IN);                          // 8-aligned; 256 % 8 == 0
    const float* src = (col < MEM_) ? (h + (size_t)row * MEM_ + col)
                                    : (x + (size_t)row * IN_DIM + (col - MEM_));
    const float4 a = ((const float4*)src)[0];
    const float4 b = ((const float4*)src)[1];
    bf16_t* d = Cb + base;
    d[0] = (bf16_t)a.x; d[1] = (bf16_t)a.y; d[2] = (bf16_t)a.z; d[3] = (bf16_t)a.w;
    d[4] = (bf16_t)b.x; d[5] = (bf16_t)b.y; d[6] = (bf16_t)b.z; d[7] = (bf16_t)b.w;
}

// WT[n][k] = [W1|Wd|Wu](k, n) as bf16; bcat[n] = [b1|bd|bu](n)
__global__ __launch_bounds__(256)
void lmc_pack_wt(const float* __restrict__ W1, const float* __restrict__ b1,
                 const float* __restrict__ Wd, const float* __restrict__ bd,
                 const float* __restrict__ Wu, const float* __restrict__ bu,
                 bf16_t* __restrict__ WT, float* __restrict__ bcat)
{
    const int gid = blockIdx.x * 256 + threadIdx.x;              // 1024 * 544
    const int n  = gid / 544;
    const int kc = gid % 544;
    const int k0 = kc * 8;
    const float* W; const float* bias; int ldw; int c;
    if (n < 512)      { W = W1; bias = b1; ldw = 512; c = n; }
    else if (n < 768) { W = Wd; bias = bd; ldw = 256; c = n - 512; }
    else              { W = Wu; bias = bu; ldw = 256; c = n - 768; }
    bf16_t* d = WT + (size_t)n * D_IN + k0;
    #pragma unroll
    for (int i = 0; i < 8; ++i) d[i] = (bf16_t)W[(size_t)(k0 + i) * ldw + c];
    if (kc == 0) bcat[n] = bias[c];
}

// W2T[n][k] = W2(k, n) as bf16
__global__ __launch_bounds__(256)
void lmc_pack_w2t(const float* __restrict__ W2, bf16_t* __restrict__ W2T)
{
    const int gid = blockIdx.x * 256 + threadIdx.x;              // 256 * 64
    const int n  = gid / 64;
    const int kc = gid % 64;
    const int k0 = kc * 8;
    bf16_t* d = W2T + (size_t)n * 512 + k0;
    #pragma unroll
    for (int i = 0; i < 8; ++i) d[i] = (bf16_t)W2[(size_t)(k0 + i) * 256 + n];
}

// ---------------------------------------------------------------------------
// Kernel A: Y[8192,1024] = Cb @ WT^T + bcat
// block = 256 threads (8 waves), tile 128(M) x 128(N) x 32(K)
// double-buffered LDS fed by the Tensor Data Mover (fallback: async b128)
// grid = (8192/128, 1024/128)
// ---------------------------------------------------------------------------
__global__ __launch_bounds__(256)
void lmc_gemm_fused(const bf16_t* __restrict__ Cb, const bf16_t* __restrict__ WT,
                    const float* __restrict__ bcat, float* __restrict__ Y)
{
    const int tid   = threadIdx.x;
    const int lane  = tid & 31;
    const int wid   = tid >> 5;
    const int lm    = lane & 15;
    const int khalf = lane >> 4;
    const int M0    = blockIdx.x * 128;
    const int N0    = blockIdx.y * 128;

    __shared__ bf16_t lds_a[2][128][40];    // [buf][m][k], 80B rows (64B data + 16B pad)
    __shared__ bf16_t lds_b[2][128][40];    // [buf][n][k]

    const int wm = wid >> 1;                // 0..3 -> 32 rows
    const int wn = wid & 1;                 // 0..1 -> 64 cols

    v8f zero = {};
    v8f acc[2][4];
    #pragma unroll
    for (int i = 0; i < 2; ++i)
        #pragma unroll
        for (int j = 0; j < 4; ++j) acc[i][j] = zero;

#ifdef USE_TDM
    const bool issuer = (wid == 0);         // one wave drives the TDM
    auto issueTile = [&](int k0, int p) {
        tdm_load_2d((unsigned)(uintptr_t)&lds_a[p][0][0],
                    Cb + (size_t)M0 * D_IN + k0,
                    /*tile*/32, 128, /*tensor*/D_IN, B_ROWS, /*stride0*/D_IN);
        tdm_load_2d((unsigned)(uintptr_t)&lds_b[p][0][0],
                    WT + (size_t)N0 * D_IN + k0,
                    /*tile*/32, 128, /*tensor*/D_IN, N_TOTAL, /*stride0*/D_IN);
    };
#else
    auto issueTile = [&](int k0, int p) {   // 512 16B-chunks per tile, 2 per thread
        #pragma unroll
        for (int q = 0; q < 2; ++q) {
            const int c = tid + q * 256;
            const int r = c >> 2;
            const int s = (c & 3) * 8;
            async_copy16(&lds_a[p][r][s], Cb + (size_t)(M0 + r) * D_IN + k0 + s);
            async_copy16(&lds_b[p][r][s], WT + (size_t)(N0 + r) * D_IN + k0 + s);
        }
    };
#endif

    const int NT = D_IN / 32;               // 136
    int p = 0;
#ifdef USE_TDM
    if (issuer) issueTile(0, 0);
#else
    issueTile(0, 0);
#endif

    for (int t = 0; t < NT; ++t) {
#ifdef USE_TDM
        if (issuer) {
            if (t + 1 < NT) { issueTile((t + 1) * 32, p ^ 1); wait_tensor_le2(); }
            else            { wait_tensor_0(); }
        }
#else
        if (t + 1 < NT) { issueTile((t + 1) * 32, p ^ 1); wait_async_le4(); }
        else            { wait_async_0(); }
#endif
        __syncthreads();                    // tile t visible to all waves

        v16bf afrag[2];
        #pragma unroll
        for (int i = 0; i < 2; ++i) {
            const int row = wm * 32 + i * 16 + lm;
            const bf16x8 lo = *(const bf16x8*)&lds_a[p][row][khalf * 8];
            const bf16x8 hi = *(const bf16x8*)&lds_a[p][row][khalf * 8 + 16];
            #pragma unroll
            for (int e = 0; e < 8; ++e) { afrag[i][e] = lo[e]; afrag[i][8 + e] = hi[e]; }
        }
        v16bf bfrag[4];
        #pragma unroll
        for (int j = 0; j < 4; ++j) {
            const int col = wn * 64 + j * 16 + lm;
            const bf16x8 lo = *(const bf16x8*)&lds_b[p][col][khalf * 8];
            const bf16x8 hi = *(const bf16x8*)&lds_b[p][col][khalf * 8 + 16];
            #pragma unroll
            for (int e = 0; e < 8; ++e) { bfrag[j][e] = lo[e]; bfrag[j][8 + e] = hi[e]; }
        }
        #pragma unroll
        for (int i = 0; i < 2; ++i)
            #pragma unroll
            for (int j = 0; j < 4; ++j)
                acc[i][j] = __builtin_amdgcn_wmma_f32_16x16x32_bf16(
                    false, afrag[i], false, bfrag[j], (short)0, acc[i][j], false, false);
        __syncthreads();                    // all waves done reading buf p
        p ^= 1;
    }

    #pragma unroll
    for (int i = 0; i < 2; ++i)
        #pragma unroll
        for (int j = 0; j < 4; ++j)
            #pragma unroll
            for (int r = 0; r < 8; ++r) {
                const int row_l = wm * 32 + i * 16 + r + khalf * 8;
                const int col_l = wn * 64 + j * 16 + lm;
                Y[(size_t)(M0 + row_l) * N_TOTAL + N0 + col_l] =
                    acc[i][j][r] + bcat[N0 + col_l];
            }
}

// ---------------------------------------------------------------------------
// Kernel B: per-row LayerNorm(512) -> exact GELU -> bf16 cast
// ---------------------------------------------------------------------------
__global__ __launch_bounds__(256)
void lmc_ln_gelu(const float* __restrict__ Y, const float* __restrict__ g1,
                 const float* __restrict__ be1, bf16_t* __restrict__ Tb)
{
    const int row = blockIdx.x;
    const int tid = threadIdx.x;
    const float* yr = Y + (size_t)row * N_TOTAL;

    const float v0 = yr[tid];
    const float v1 = yr[tid + 256];

    __shared__ float s_sum[256];
    __shared__ float s_sq[256];
    s_sum[tid] = v0 + v1;
    s_sq[tid]  = v0 * v0 + v1 * v1;
    __syncthreads();
    #pragma unroll
    for (int off = 128; off > 0; off >>= 1) {
        if (tid < off) { s_sum[tid] += s_sum[tid + off]; s_sq[tid] += s_sq[tid + off]; }
        __syncthreads();
    }
    const float mu   = s_sum[0] * (1.0f / 512.0f);
    const float var  = s_sq[0] * (1.0f / 512.0f) - mu * mu;
    const float rstd = rsqrtf(var + 1e-5f);

    #pragma unroll
    for (int e = 0; e < 2; ++e) {
        const int   c = tid + e * 256;
        const float v = (e == 0) ? v0 : v1;
        const float t = (v - mu) * rstd * g1[c] + be1[c];
        const float g = 0.5f * t * (1.0f + erff(t * 0.70710678118654752f));
        Tb[(size_t)row * 512 + c] = (bf16_t)g;
    }
}

// ---------------------------------------------------------------------------
// Kernel C: h_cand = tanh(Tb @ W2T^T + b2); gates from Y; decay mix; L2 renorm
// block = 256 threads (8 waves), tile 64(M) x 256(N=full) x 32(K), async+dbuf
// ---------------------------------------------------------------------------
__global__ __launch_bounds__(256)
void lmc_gemm2_epilogue(const bf16_t* __restrict__ Tb, const bf16_t* __restrict__ W2T,
                        const float* __restrict__ b2, const float* __restrict__ Y,
                        const float* __restrict__ h,  float* __restrict__ out)
{
    const int tid   = threadIdx.x;
    const int lane  = tid & 31;
    const int wid   = tid >> 5;
    const int lm    = lane & 15;
    const int khalf = lane >> 4;
    const int M0    = blockIdx.x * 64;
    const int wm    = wid >> 2;             // 0..1 -> 32 rows
    const int wn    = wid & 3;              // 0..3 -> 64 cols

    __shared__ bf16_t lds_a[2][64][40];
    __shared__ bf16_t lds_b[2][256][40];
    __shared__ float  s_nh2[64];
    __shared__ float  s_h2[64];
    __shared__ float  s_scale[64];

    v8f zero = {};
    v8f acc[2][4];
    #pragma unroll
    for (int i = 0; i < 2; ++i)
        #pragma unroll
        for (int j = 0; j < 4; ++j) acc[i][j] = zero;

    auto issueTile = [&](int k0, int p) {
        {   // A: 256 chunks, 1 per thread
            const int r = tid >> 2;
            const int s = (tid & 3) * 8;
            async_copy16(&lds_a[p][r][s], Tb + (size_t)(M0 + r) * 512 + k0 + s);
        }
        #pragma unroll
        for (int q = 0; q < 4; ++q) {       // B: 1024 chunks, 4 per thread
            const int c = tid + q * 256;
            const int r = c >> 2;
            const int s = (c & 3) * 8;
            async_copy16(&lds_b[p][r][s], W2T + (size_t)r * 512 + k0 + s);
        }
    };

    const int NT = 512 / 32;                // 16
    int p = 0;
    issueTile(0, 0);

    for (int t = 0; t < NT; ++t) {
        if (t + 1 < NT) { issueTile((t + 1) * 32, p ^ 1); wait_async_le5(); }
        else            { wait_async_0(); }
        __syncthreads();

        v16bf afrag[2];
        #pragma unroll
        for (int i = 0; i < 2; ++i) {
            const int row = wm * 32 + i * 16 + lm;
            const bf16x8 lo = *(const bf16x8*)&lds_a[p][row][khalf * 8];
            const bf16x8 hi = *(const bf16x8*)&lds_a[p][row][khalf * 8 + 16];
            #pragma unroll
            for (int e = 0; e < 8; ++e) { afrag[i][e] = lo[e]; afrag[i][8 + e] = hi[e]; }
        }
        v16bf bfrag[4];
        #pragma unroll
        for (int j = 0; j < 4; ++j) {
            const int col = wn * 64 + j * 16 + lm;
            const bf16x8 lo = *(const bf16x8*)&lds_b[p][col][khalf * 8];
            const bf16x8 hi = *(const bf16x8*)&lds_b[p][col][khalf * 8 + 16];
            #pragma unroll
            for (int e = 0; e < 8; ++e) { bfrag[j][e] = lo[e]; bfrag[j][8 + e] = hi[e]; }
        }
        #pragma unroll
        for (int i = 0; i < 2; ++i)
            #pragma unroll
            for (int j = 0; j < 4; ++j)
                acc[i][j] = __builtin_amdgcn_wmma_f32_16x16x32_bf16(
                    false, afrag[i], false, bfrag[j], (short)0, acc[i][j], false, false);
        __syncthreads();
        p ^= 1;
    }

    if (tid < 64) { s_nh2[tid] = 0.0f; s_h2[tid] = 0.0f; }
    __syncthreads();

    #pragma unroll
    for (int i = 0; i < 2; ++i)
        #pragma unroll
        for (int j = 0; j < 4; ++j)
            #pragma unroll
            for (int r = 0; r < 8; ++r) {
                const int    row_l = wm * 32 + i * 16 + r + khalf * 8;
                const int    col   = wn * 64 + j * 16 + lm;
                const size_t row   = (size_t)(M0 + row_l);
                const float hc  = tanhf(acc[i][j][r] + b2[col]);
                const float dl  = Y[row * N_TOTAL + 512 + col];
                const float ul  = Y[row * N_TOTAL + 768 + col];
                const float dec = 0.1f + 9.9f * sigm_(dl) + 1e-6f;
                const float u   = sigm_(ul);
                const float hv  = h[row * MEM_ + col];
                const float nh  = (1.0f - u) * (hv / dec) + u * hc;
                acc[i][j][r] = nh;
                atomicAdd(&s_nh2[row_l], nh * nh);                 // ds_add_f32
                atomicAdd(&s_h2[row_l],  hv * hv);
            }
    __syncthreads();

    if (tid < 64) {
        const float hn = fmaxf(sqrtf(s_h2[tid]), 1e-6f);
        const float nn = fmaxf(sqrtf(s_nh2[tid]), 1e-12f);
        s_scale[tid] = hn / nn;
    }
    __syncthreads();

    #pragma unroll
    for (int i = 0; i < 2; ++i)
        #pragma unroll
        for (int j = 0; j < 4; ++j)
            #pragma unroll
            for (int r = 0; r < 8; ++r) {
                const int row_l = wm * 32 + i * 16 + r + khalf * 8;
                const int col   = wn * 64 + j * 16 + lm;
                out[(size_t)(M0 + row_l) * MEM_ + col] = acc[i][j][r] * s_scale[row_l];
            }
}

// ---------------------------------------------------------------------------
extern "C" void kernel_launch(void* const* d_in, const int* in_sizes, int n_in,
                              void* d_out, int out_size, void* d_ws, size_t ws_size,
                              hipStream_t stream)
{
    const float* x   = (const float*)d_in[0];
    const float* h   = (const float*)d_in[1];
    const float* W1  = (const float*)d_in[2];
    const float* b1  = (const float*)d_in[3];
    const float* g1  = (const float*)d_in[4];
    const float* be1 = (const float*)d_in[5];
    const float* W2  = (const float*)d_in[6];
    const float* b2  = (const float*)d_in[7];
    const float* Wd  = (const float*)d_in[8];
    const float* bd  = (const float*)d_in[9];
    const float* Wu  = (const float*)d_in[10];
    const float* bu  = (const float*)d_in[11];
    float* out = (float*)d_out;

    char* ws = (char*)d_ws;
    float*  Y    = (float*) (ws);                                         // 33,554,432 B
    bf16_t* Tb   = (bf16_t*)(ws + 33554432);                              //  8,388,608 B
    bf16_t* Cb   = (bf16_t*)(ws + 41943040);                              // 71,303,168 B
    bf16_t* WT   = (bf16_t*)(ws + 113246208);                             //  8,912,896 B
    bf16_t* W2T  = (bf16_t*)(ws + 122159104);                             //    262,144 B
    float*  bcat = (float*) (ws + 122421248);                             //      4,096 B

    lmc_pack_combined<<<(B_ROWS * (D_IN / 8)) / 256, 256, 0, stream>>>(x, h, Cb);
    lmc_pack_wt<<<(N_TOTAL * (D_IN / 8)) / 256, 256, 0, stream>>>(W1, b1, Wd, bd, Wu, bu, WT, bcat);
    lmc_pack_w2t<<<(256 * (512 / 8)) / 256, 256, 0, stream>>>(W2, W2T);

    dim3 gA(B_ROWS / 128, N_TOTAL / 128);
    lmc_gemm_fused<<<gA, 256, 0, stream>>>(Cb, WT, bcat, Y);
    lmc_ln_gelu<<<B_ROWS, 256, 0, stream>>>(Y, g1, be1, Tb);
    lmc_gemm2_epilogue<<<B_ROWS / 64, 256, 0, stream>>>(Tb, W2T, b2, Y, h, out);
}
